// TSUBridge_2482491097796
// MI455X (gfx1250) — matrix operand order repaired and checked by reading, not compiled, not run on hardware
//
#include <hip/hip_runtime.h>

// ---------------------------------------------------------------------------
// TSU settling bridge on MI455X (gfx1250):
//   x_{t+1} = x_t + dt * ( tanh(x_t@W1 + b1) @ W2 + b2 ),  20 steps.
// ~1.37 TFLOP of f16 WMMA GEMM; full working set (~72MB) is L2-resident, so
// the kernel is matrix-unit bound. Double-buffered LDS tiles with CDNA5
// async global->LDS copies (ASYNCcnt) when the toolchain exposes them.
// ---------------------------------------------------------------------------

typedef __attribute__((ext_vector_type(16))) _Float16 v16h;
typedef __attribute__((ext_vector_type(8)))  _Float16 v8h;
typedef __attribute__((ext_vector_type(8)))  float    v8f;
typedef __attribute__((ext_vector_type(4)))  int      i4;

#define BATCH  4096
#define DIM_   1024
#define HID_   4096
#define NSTEPS 20
#define DT     (1.0f / 20.0f)

// Block = 256 threads = 8 wave32; block tile 128x128; wave tile 32x64
// (2x4 WMMA 16x16 tiles); K slab = 32 (one WMMA K), double buffered.
#define BM   128
#define BN   128
#define KT   32
#define LDSW 40   // halves per LDS row: 32 data + 8 pad -> 80B, 16B aligned,
                  // bank stride 20 -> 16 consecutive rows hit distinct banks.

#if __has_builtin(__builtin_amdgcn_global_load_async_to_lds_b128) && \
    __has_builtin(__builtin_amdgcn_s_wait_asynccnt)
#define USE_ASYNC_COPY 1
// Builtin signature (from clang diagnostic): param0 = AS1 int4* (global src),
// param1 = AS3 int4* (LDS dst), then imm offset, imm cpol.
typedef __attribute__((address_space(1))) i4* gi4p;
typedef __attribute__((address_space(3))) i4* li4p;
// Global AS pointer == flat pointer value for global memory.
__device__ __forceinline__ gi4p to_glb(const void* p) {
  return (gi4p)(unsigned long long)p;
}
// Flat LDS pointer carries the LDS byte offset in its low 32 bits
// (ISA 10.2), so trunc-cast yields the AS3 offset.
__device__ __forceinline__ li4p to_lds(void* p) {
  return (li4p)(unsigned int)(unsigned long long)p;
}
#else
#define USE_ASYNC_COPY 0
#endif

// ---- fragment loaders (layouts per CDNA5 ISA 7.12.2) ----------------------
// A (16x32 f16, MxK): lanes 0-15 hold row m, K {0..7,16..23};
//                     lanes 16-31 hold row m, K {8..15,24..31}.
__device__ __forceinline__ v16h frag_a(const _Float16* base, int row0, int lane) {
  const int r  = row0 + (lane & 15);
  const int kb = (lane < 16) ? 0 : 8;
  const _Float16* p = base + r * LDSW + kb;
  v8h lo = *(const v8h*)(p);
  v8h hi = *(const v8h*)(p + 16);
  return __builtin_shufflevector(lo, hi, 0,1,2,3,4,5,6,7,8,9,10,11,12,13,14,15);
}
// B (32x16 f16, KxN) stored N-major in LDS (row = n, contiguous K):
// lane holds column n = lane%16; lanes 0-15 K {0..7,16..23}, lanes 16-31 K {8..15,24..31}.
__device__ __forceinline__ v16h frag_b(const _Float16* base, int row0, int lane) {
  const int r  = row0 + (lane & 15);
  const int kb = (lane < 16) ? 0 : 8;
  const _Float16* p = base + r * LDSW + kb;
  v8h lo = *(const v8h*)(p);
  v8h hi = *(const v8h*)(p + 16);
  return __builtin_shufflevector(lo, hi, 0,1,2,3,4,5,6,7,8,9,10,11,12,13,14,15);
}

// ---- double-buffered WMMA mainloop: C[128x128] += A[128xK] * Bt[128xK]^T --
// A row-major [M x K], Bt row-major [N x K] (B pre-transposed).
__device__ __forceinline__ void mma_block(
    const _Float16* __restrict__ A, const _Float16* __restrict__ Bt,
    int K, _Float16* sA0, _Float16* sA1, _Float16* sB0, _Float16* sB1,
    v8f acc[2][4])
{
  const int tid  = threadIdx.x;
  const int lane = tid & 31;
  const int wave = tid >> 5;
  const int wm   = (wave & 3) << 5;
  const int wn   = (wave >> 2) << 6;
  const size_t row0 = (size_t)blockIdx.y * BM;
  const size_t col0 = (size_t)blockIdx.x * BN;

  // Per-thread staging coords: two 16B chunks per tile buffer.
  const int r0 = tid >> 2,         s0 = (tid & 3) << 3;
  const int r1 = (tid + 256) >> 2, s1 = ((tid + 256) & 3) << 3;
  const _Float16* gA0 = &A [(row0 + r0) * (size_t)K + s0];
  const _Float16* gA1 = &A [(row0 + r1) * (size_t)K + s1];
  const _Float16* gB0 = &Bt[(col0 + r0) * (size_t)K + s0];
  const _Float16* gB1 = &Bt[(col0 + r1) * (size_t)K + s1];
  const int lA0 = r0 * LDSW + s0, lA1 = r1 * LDSW + s1;

  const int nk = K / KT;

#if USE_ASYNC_COPY
  // CDNA5 async global->LDS: no VGPR staging, tracked by ASYNCcnt.
  auto stage = [&](int t, _Float16* dA, _Float16* dB) {
    const int k0 = t * KT;
    __builtin_amdgcn_global_load_async_to_lds_b128(to_glb(gA0 + k0), to_lds(&dA[lA0]), 0, 0);
    __builtin_amdgcn_global_load_async_to_lds_b128(to_glb(gA1 + k0), to_lds(&dA[lA1]), 0, 0);
    __builtin_amdgcn_global_load_async_to_lds_b128(to_glb(gB0 + k0), to_lds(&dB[lA0]), 0, 0);
    __builtin_amdgcn_global_load_async_to_lds_b128(to_glb(gB1 + k0), to_lds(&dB[lA1]), 0, 0);
  };
  stage(0, sA0, sB0);
  __builtin_amdgcn_s_wait_asynccnt(0);
  __syncthreads();
  for (int t = 0; t < nk; ++t) {
    _Float16* cA = (t & 1) ? sA1 : sA0;
    _Float16* cB = (t & 1) ? sB1 : sB0;
    if (t + 1 < nk) stage(t + 1, (t & 1) ? sA0 : sA1, (t & 1) ? sB0 : sB1);

    v16h af[2], bf[4];
    #pragma unroll
    for (int m = 0; m < 2; ++m) af[m] = frag_a(cA, wm + (m << 4), lane);
    #pragma unroll
    for (int n = 0; n < 4; ++n) bf[n] = frag_b(cB, wn + (n << 4), lane);
    #pragma unroll
    for (int m = 0; m < 2; ++m)
      #pragma unroll
      for (int n = 0; n < 4; ++n)
        acc[m][n] = __builtin_amdgcn_wmma_f32_16x16x32_f16(
            false, af[m], false, bf[n], (short)0, acc[m][n], false, false);

    __builtin_amdgcn_s_wait_asynccnt(0);
    __syncthreads();
  }
#else
  // Fallback: global->VGPR prefetch of slab t+1 overlapped with WMMAs on t.
  v8h ra0, ra1, rb0, rb1;
  auto gload = [&](int t) {
    const int k0 = t * KT;
    ra0 = *(const v8h*)(gA0 + k0);
    ra1 = *(const v8h*)(gA1 + k0);
    rb0 = *(const v8h*)(gB0 + k0);
    rb1 = *(const v8h*)(gB1 + k0);
  };
  auto sstore = [&](_Float16* dA, _Float16* dB) {
    *(v8h*)&dA[lA0] = ra0;
    *(v8h*)&dA[lA1] = ra1;
    *(v8h*)&dB[lA0] = rb0;
    *(v8h*)&dB[lA1] = rb1;
  };
  gload(0);
  sstore(sA0, sB0);
  __syncthreads();
  for (int t = 0; t < nk; ++t) {
    _Float16* cA = (t & 1) ? sA1 : sA0;
    _Float16* cB = (t & 1) ? sB1 : sB0;
    if (t + 1 < nk) gload(t + 1);

    v16h af[2], bf[4];
    #pragma unroll
    for (int m = 0; m < 2; ++m) af[m] = frag_a(cA, wm + (m << 4), lane);
    #pragma unroll
    for (int n = 0; n < 4; ++n) bf[n] = frag_b(cB, wn + (n << 4), lane);
    #pragma unroll
    for (int m = 0; m < 2; ++m)
      #pragma unroll
      for (int n = 0; n < 4; ++n)
        acc[m][n] = __builtin_amdgcn_wmma_f32_16x16x32_f16(
            false, af[m], false, bf[n], (short)0, acc[m][n], false, false);

    if (t + 1 < nk) sstore((t & 1) ? sA0 : sA1, (t & 1) ? sB0 : sB1);
    __syncthreads();
  }
#endif
}

// ---- GEMM1: H = tanh(X @ W1 + b1), H stored as f16 ------------------------
__global__ __launch_bounds__(256)
void gemm1_tanh_kernel(const _Float16* __restrict__ X,   // [BATCH x DIM] f16
                       const _Float16* __restrict__ W1t, // [HID x DIM]  f16 (W1^T)
                       const float*    __restrict__ b1,  // [HID]
                       _Float16*       __restrict__ H)   // [BATCH x HID] f16
{
  __shared__ _Float16 sA[2][BM * LDSW];
  __shared__ _Float16 sB[2][BN * LDSW];
  v8f acc[2][4] = {};
  mma_block(X, W1t, DIM_, sA[0], sA[1], sB[0], sB[1], acc);

  const int lane = threadIdx.x & 31;
  const int wave = threadIdx.x >> 5;
  // C/D layout: VGPR j -> row (j + 8*(lane>=16)), col = lane%16.
  const int mb = (int)blockIdx.y * BM + ((wave & 3) << 5) + ((lane < 16) ? 0 : 8);
  const int nb = (int)blockIdx.x * BN + ((wave >> 2) << 6) + (lane & 15);
  #pragma unroll
  for (int m = 0; m < 2; ++m) {
    #pragma unroll
    for (int n = 0; n < 4; ++n) {
      const int col  = nb + (n << 4);
      const float bv = b1[col];
      const int mrow = mb + (m << 4);
      #pragma unroll
      for (int j = 0; j < 8; ++j) {
        const float v = acc[m][n][j] + bv;
        // tanh(v) = 1 - 2/(exp(2v)+1): overflow-safe (inf -> 1, 0 -> -1).
        const float e = __expf(2.0f * v);
        const float t = 1.0f - 2.0f / (e + 1.0f);
        H[(size_t)(mrow + j) * HID_ + col] = (_Float16)t;
      }
    }
  }
}

// ---- GEMM2: Xout = Xin + dt*(H @ W2 + b2); also emit f16 copy of Xout -----
__global__ __launch_bounds__(256)
void gemm2_step_kernel(const _Float16* __restrict__ H,   // [BATCH x HID] f16
                       const _Float16* __restrict__ W2t, // [DIM x HID]  f16 (W2^T)
                       const float*    __restrict__ b2,  // [DIM]
                       const float*    __restrict__ Xin, // [BATCH x DIM] f32
                       float*          __restrict__ Xout,// [BATCH x DIM] f32
                       _Float16*       __restrict__ Xh)  // [BATCH x DIM] f16
{
  __shared__ _Float16 sA[2][BM * LDSW];
  __shared__ _Float16 sB[2][BN * LDSW];
  v8f acc[2][4] = {};
  mma_block(H, W2t, HID_, sA[0], sA[1], sB[0], sB[1], acc);

  const int lane = threadIdx.x & 31;
  const int wave = threadIdx.x >> 5;
  const int mb = (int)blockIdx.y * BM + ((wave & 3) << 5) + ((lane < 16) ? 0 : 8);
  const int nb = (int)blockIdx.x * BN + ((wave >> 2) << 6) + (lane & 15);
  #pragma unroll
  for (int m = 0; m < 2; ++m) {
    #pragma unroll
    for (int n = 0; n < 4; ++n) {
      const int col  = nb + (n << 4);
      const float bv = b2[col];
      const int mrow = mb + (m << 4);
      #pragma unroll
      for (int j = 0; j < 8; ++j) {
        const size_t idx = (size_t)(mrow + j) * DIM_ + col;
        const float xn = Xin[idx] + DT * (acc[m][n][j] + bv);
        Xout[idx] = xn;
        Xh[idx]   = (_Float16)xn;
      }
    }
  }
}

// ---- prep: fp32 weight [R x C] -> f16 transposed [C x R] ------------------
__global__ void transpose_f32_to_f16(const float* __restrict__ W,
                                     _Float16* __restrict__ Wt, int R, int C) {
  const size_t o = (size_t)blockIdx.x * blockDim.x + threadIdx.x;
  if (o >= (size_t)R * C) return;
  const int k = (int)(o % (size_t)R);
  const int n = (int)(o / (size_t)R);
  Wt[o] = (_Float16)W[(size_t)k * C + n];
}

__global__ void f32_to_f16(const float* __restrict__ src,
                           _Float16* __restrict__ dst, int n) {
  const int i = blockIdx.x * blockDim.x + threadIdx.x;
  if (i < n) dst[i] = (_Float16)src[i];
}

// ---------------------------------------------------------------------------
extern "C" void kernel_launch(void* const* d_in, const int* in_sizes, int n_in,
                              void* d_out, int out_size, void* d_ws, size_t ws_size,
                              hipStream_t stream) {
  (void)in_sizes; (void)n_in; (void)out_size; (void)ws_size;
  const float* x0 = (const float*)d_in[0];
  const float* W1 = (const float*)d_in[1];
  const float* b1 = (const float*)d_in[2];
  const float* W2 = (const float*)d_in[3];
  const float* b2 = (const float*)d_in[4];
  float* out = (float*)d_out;

  // Workspace layout (72 MB total):
  const size_t MB = 1024ull * 1024ull;
  char* ws = (char*)d_ws;
  _Float16* W1t = (_Float16*)(ws + 0 * MB);   // [HID x DIM] f16   (8 MB)
  _Float16* W2t = (_Float16*)(ws + 8 * MB);   // [DIM x HID] f16   (8 MB)
  _Float16* Xh  = (_Float16*)(ws + 16 * MB);  // [BATCH x DIM] f16 (8 MB)
  _Float16* Hh  = (_Float16*)(ws + 24 * MB);  // [BATCH x HID] f16 (32 MB)
  float*    Xf  = (float*)   (ws + 56 * MB);  // [BATCH x DIM] f32 (16 MB)

  // One-time per call: f16 weights (pre-transposed to N-major) + f16 x0.
  {
    const size_t nW = (size_t)DIM_ * HID_;
    const int blocks = (int)((nW + 255) / 256);
    transpose_f32_to_f16<<<blocks, 256, 0, stream>>>(W1, W1t, DIM_, HID_);
    transpose_f32_to_f16<<<blocks, 256, 0, stream>>>(W2, W2t, HID_, DIM_);
    const int nX = BATCH * DIM_;
    f32_to_f16<<<(nX + 255) / 256, 256, 0, stream>>>(x0, Xh, nX);
  }

  const dim3 block(256);
  const dim3 grid1(HID_ / BN, BATCH / BM);  // 32 x 32 blocks
  const dim3 grid2(DIM_ / BN, BATCH / BM);  //  8 x 32 blocks

  for (int s = 0; s < NSTEPS; ++s) {
    const float* xin  = (s == 0) ? x0 : Xf;
    float*       xout = (s == NSTEPS - 1) ? out : Xf;
    gemm1_tanh_kernel<<<grid1, block, 0, stream>>>(Xh, W1t, b1, Hh);
    gemm2_step_kernel<<<grid2, block, 0, stream>>>(Hh, W2t, b2, xin, xout, Xh);
  }
}
